// VisionBlock_72859825210074
// MI455X (gfx1250) — matrix operand order repaired and checked by reading, not compiled, not run on hardware
//
#include <hip/hip_runtime.h>
#include <hip/hip_bf16.h>
#include <math.h>

#define C_DIM 1152
#define NTOK  16384
#define HEADS 16
#define HDIM  72
#define INNER_DIM 4608
#define N3C   3456
#define SEQ   1024

typedef __attribute__((ext_vector_type(16))) __bf16 v16bf;
typedef __attribute__((ext_vector_type(8)))  float  v8f;
typedef unsigned short ushort_t;
typedef unsigned int   uint_t;

union bfu { unsigned short u; __bf16 b; };

__device__ __forceinline__ __bf16 asbf(unsigned short u) { bfu x; x.u = u; return x.b; }

__device__ __forceinline__ unsigned short f2bf(float f) {
    unsigned int u = __float_as_uint(f);
    u += 0x7FFFu + ((u >> 16) & 1u);   // round-to-nearest-even
    return (unsigned short)(u >> 16);
}
__device__ __forceinline__ v8f zero8() {
    v8f z;
#pragma unroll
    for (int i = 0; i < 8; ++i) z[i] = 0.0f;
    return z;
}

// CDNA5 async global->LDS copy (16B per lane), tracked by ASYNCcnt.
// Flat pointers to __shared__ carry the LDS offset in their low 32 bits.
__device__ __forceinline__ void async_copy_b128(void* lds_ptr, const void* gptr) {
    unsigned lds = (unsigned)(uintptr_t)lds_ptr;
    unsigned long long ga = (unsigned long long)(uintptr_t)gptr;
    asm volatile("global_load_async_to_lds_b128 %0, %1, off"
                 :: "v"(lds), "v"(ga) : "memory");
}
__device__ __forceinline__ void async_wait0() {
    asm volatile("s_wait_asynccnt 0" ::: "memory");
}

// DPP butterfly reductions within each 16-lane half of a wave32.
__device__ __forceinline__ float dpp_sum16(float x) {
    int t;
    t = __builtin_amdgcn_update_dpp(0, __float_as_int(x), 0xB1, 0xF, 0xF, true);
    x += __int_as_float(t);
    t = __builtin_amdgcn_update_dpp(0, __float_as_int(x), 0x4E, 0xF, 0xF, true);
    x += __int_as_float(t);
    t = __builtin_amdgcn_update_dpp(0, __float_as_int(x), 0x141, 0xF, 0xF, true);
    x += __int_as_float(t);
    t = __builtin_amdgcn_update_dpp(0, __float_as_int(x), 0x140, 0xF, 0xF, true);
    x += __int_as_float(t);
    return x;
}
__device__ __forceinline__ float dpp_max16(float x) {
    int t;
    t = __builtin_amdgcn_update_dpp(0, __float_as_int(x), 0xB1, 0xF, 0xF, true);
    x = fmaxf(x, __int_as_float(t));
    t = __builtin_amdgcn_update_dpp(0, __float_as_int(x), 0x4E, 0xF, 0xF, true);
    x = fmaxf(x, __int_as_float(t));
    t = __builtin_amdgcn_update_dpp(0, __float_as_int(x), 0x141, 0xF, 0xF, true);
    x = fmaxf(x, __int_as_float(t));
    t = __builtin_amdgcn_update_dpp(0, __float_as_int(x), 0x140, 0xF, 0xF, true);
    x = fmaxf(x, __int_as_float(t));
    return x;
}

// ---------------------------------------------------------------------------
// LayerNorm (fp32 in) -> bf16 out, one block per row of length C_DIM
// ---------------------------------------------------------------------------
__global__ __launch_bounds__(256)
void ln_kernel(const float* __restrict__ x, const float* __restrict__ g,
               const float* __restrict__ b, ushort_t* __restrict__ out) {
    __shared__ float red[16];
    const int row = blockIdx.x;
    const float4* xr4 = (const float4*)(x + (size_t)row * C_DIM);
    const float4* g4  = (const float4*)g;
    const float4* b4  = (const float4*)b;
    float s = 0.0f, s2 = 0.0f;
    for (int i = threadIdx.x; i < C_DIM / 4; i += 256) {
        float4 v = xr4[i];
        s  += v.x + v.y + v.z + v.w;
        s2 += v.x * v.x + v.y * v.y + v.z * v.z + v.w * v.w;
    }
#pragma unroll
    for (int off = 16; off > 0; off >>= 1) {
        s  += __shfl_down(s,  off, 32);
        s2 += __shfl_down(s2, off, 32);
    }
    const int wid = threadIdx.x >> 5;
    if ((threadIdx.x & 31) == 0) { red[wid] = s; red[8 + wid] = s2; }
    __syncthreads();
    if (threadIdx.x == 0) {
        float ts = 0.0f, ts2 = 0.0f;
        for (int i = 0; i < 8; ++i) { ts += red[i]; ts2 += red[8 + i]; }
        float mu  = ts / (float)C_DIM;
        float var = ts2 / (float)C_DIM - mu * mu;
        red[0] = mu;
        red[1] = rsqrtf(var + 1e-6f);
    }
    __syncthreads();
    const float mu = red[0], rstd = red[1];
    uint2* orow = (uint2*)(out + (size_t)row * C_DIM);
    for (int i = threadIdx.x; i < C_DIM / 4; i += 256) {
        float4 v = xr4[i], gg = g4[i], bb = b4[i];
        unsigned int o0 = f2bf((v.x - mu) * rstd * gg.x + bb.x);
        unsigned int o1 = f2bf((v.y - mu) * rstd * gg.y + bb.y);
        unsigned int o2 = f2bf((v.z - mu) * rstd * gg.z + bb.z);
        unsigned int o3 = f2bf((v.w - mu) * rstd * gg.w + bb.w);
        uint2 pk; pk.x = o0 | (o1 << 16); pk.y = o2 | (o3 << 16);
        orow[i] = pk;
    }
}

// ---------------------------------------------------------------------------
// fp32 -> bf16 conversion (weights); n4 = element count / 4
// ---------------------------------------------------------------------------
__global__ __launch_bounds__(256)
void cvt_kernel(const float* __restrict__ in, ushort_t* __restrict__ out, int n4) {
    int i = blockIdx.x * 256 + threadIdx.x;
    if (i < n4) {
        float4 v = ((const float4*)in)[i];
        uint2 pk;
        pk.x = (unsigned int)f2bf(v.x) | ((unsigned int)f2bf(v.y) << 16);
        pk.y = (unsigned int)f2bf(v.z) | ((unsigned int)f2bf(v.w) << 16);
        ((uint2*)out)[i] = pk;
    }
}

// ---------------------------------------------------------------------------
// WMMA GEMM: out = epilogue(A_bf16[M,K] @ W_bf16[K,N] + bias)
// block = 128 threads (4 waves), tile 64x64, K step 32.
// A tile staged via async global->LDS b128; W tile staged transposed so B
// fragments are contiguous (ds_load_b128).
// MODE 0: fp32 out = acc + bias + res          (proj, fc2)
// MODE 1: bf16 out = gelu(acc + bias)          (fc1)
// MODE 2: bf16 scatter to Q/K/V (B,H,N,D)      (qkv)
// ---------------------------------------------------------------------------
template <int MODE>
__global__ __launch_bounds__(128)
void gemm_wmma(const ushort_t* __restrict__ A, const ushort_t* __restrict__ W,
               const float* __restrict__ bias, const float* __restrict__ res,
               float* __restrict__ out_f32, ushort_t* __restrict__ out_bf16,
               ushort_t* __restrict__ qb, ushort_t* __restrict__ kb,
               ushort_t* __restrict__ vb, int K, int N) {
    __shared__ ushort_t sA[64][32];    // [row][k]
    __shared__ ushort_t sBt[64][32];   // [col][k]  (transposed)

    const int bm = blockIdx.x, bn = blockIdx.y;
    const int tid  = threadIdx.x;
    const int wave = tid >> 5;
    const int lane = tid & 31;
    const int r    = lane & 15;
    const int half = lane >> 4;

    const uint_t* Wg = (const uint_t*)W;
    const int Nd = N >> 1;

    v8f acc[4];
#pragma unroll
    for (int t = 0; t < 4; ++t) acc[t] = zero8();

    for (int k0 = 0; k0 < K; k0 += 32) {
        // stage A tile: 64 rows x 64B = 256 x 16B async chunks
#pragma unroll
        for (int it = 0; it < 2; ++it) {
            int idx = tid + it * 128;
            int rr = idx >> 2, cc = (idx & 3) * 8;
            async_copy_b128(&sA[rr][cc],
                            A + (size_t)(bm * 64 + rr) * K + k0 + cc);
        }
        // stage W tile transposed (coalesced dword reads, 2x u16 LDS writes)
#pragma unroll
        for (int it = 0; it < 8; ++it) {
            int idx = tid + it * 128;               // 32 k * 32 col-pairs
            int kk = idx >> 5, c = idx & 31;
            uint_t val = Wg[(size_t)(k0 + kk) * Nd + bn * 32 + c];
            sBt[2 * c][kk]     = (ushort_t)(val & 0xFFFFu);
            sBt[2 * c + 1][kk] = (ushort_t)(val >> 16);
        }
        if (k0 + 32 < K) {
            __builtin_prefetch(&Wg[(size_t)(k0 + 32 + (tid >> 5)) * Nd + bn * 32], 0, 1);
        }
        async_wait0();
        __syncthreads();

        // A fragment: contiguous K {0..15} / {16..31} -> two b128 runs
        v16bf af;
        const ushort_t* sArow = &sA[wave * 16 + r][0];
#pragma unroll
        for (int j = 0; j < 16; ++j) {
            int kl = half * 16 + j;
            af[j] = asbf(sArow[kl]);
        }
#pragma unroll
        for (int t = 0; t < 4; ++t) {
            v16bf bfr;
            const ushort_t* sBrow = &sBt[t * 16 + r][half * 16];
#pragma unroll
            for (int j = 0; j < 16; ++j)
                bfr[j] = asbf(sBrow[j]);
            acc[t] = __builtin_amdgcn_wmma_f32_16x16x32_bf16(
                false, af, false, bfr, (short)0, acc[t], false, false);
        }
        __syncthreads();
    }

    // epilogue: element v of acc[t] is row (v + 8*half), col (t*16 + r) of tile
#pragma unroll
    for (int t = 0; t < 4; ++t) {
#pragma unroll
        for (int v = 0; v < 8; ++v) {
            const int row = bm * 64 + wave * 16 + v + 8 * half;
            const int col = bn * 64 + t * 16 + r;
            float val = acc[t][v] + bias[col];
            if (MODE == 0) {
                size_t o = (size_t)row * N + col;
                out_f32[o] = val + res[o];
            } else if (MODE == 1) {
                float gx = 0.5f * val * (1.0f + erff(val * 0.70710678f));
                out_bf16[(size_t)row * N + col] = f2bf(gx);
            } else {  // QKV scatter
                int tsel = col / C_DIM;
                int rem  = col - tsel * C_DIM;
                int h    = rem / HDIM;
                int d    = rem - h * HDIM;
                int bidx = row >> 10;
                int nidx = row & 1023;
                ushort_t* dst = (tsel == 0) ? qb : (tsel == 1) ? kb : vb;
                dst[(((size_t)bidx * HEADS + h) * SEQ + nidx) * HDIM + d] = f2bf(val);
            }
        }
    }
}

// ---------------------------------------------------------------------------
// Flash attention. Block = 128 threads (4 waves) = 64 query rows per (b,h).
// Q/K tiles staged via async global->LDS b128 (pad columns zeroed once and
// never rewritten); V tile transposed manually. Online softmax with DPP
// butterflies (per-16-lane-half rows of the C/D layout).
// ---------------------------------------------------------------------------
__global__ __launch_bounds__(128)
void attn_kernel(const ushort_t* __restrict__ Q, const ushort_t* __restrict__ Km,
                 const ushort_t* __restrict__ V, ushort_t* __restrict__ O) {
    __shared__ ushort_t sQ[64][96];      // 12 KB
    __shared__ ushort_t sK[32][96];      // 6 KB
    __shared__ ushort_t sVt[80][32];     // 5 KB [d][key]
    __shared__ ushort_t sP[4][16][32];   // 4 KB per-wave P tiles

    const int qb   = blockIdx.x;   // 0..15 (64 query rows each)
    const int bh   = blockIdx.y;   // 0..255
    const int tid  = threadIdx.x;
    const int wave = tid >> 5;
    const int lane = tid & 31;
    const int r    = lane & 15;
    const int half = lane >> 4;
    const int q0   = qb * 64;
    const size_t base = (size_t)bh * SEQ * HDIM;
    const float scale = 0.11785113f;  // 1/sqrt(72)

    const ushort_t* Qg = Q + base;   // row stride 72 u16 = 144 B (16B aligned)
    const ushort_t* Kg = Km + base;
    const uint_t*   Vg = (const uint_t*)(V + base);

    // zero pad columns (written once; async staging never touches them)
    for (int idx = tid; idx < 64 * 24; idx += 128)
        sQ[idx / 24][72 + idx % 24] = 0;
    for (int idx = tid; idx < 32 * 24; idx += 128)
        sK[idx / 24][72 + idx % 24] = 0;
    for (int idx = tid; idx < 8 * 32; idx += 128)
        sVt[72 + (idx >> 5)][idx & 31] = 0;

    // stage Q via async: 64 rows x 9 x 16B chunks
    for (int idx = tid; idx < 64 * 9; idx += 128) {
        int rr = idx / 9, cc = (idx % 9) * 8;
        async_copy_b128(&sQ[rr][cc], Qg + (size_t)(q0 + rr) * HDIM + cc);
    }
    async_wait0();
    __syncthreads();

    // Q fragments for 3 K-chunks (contiguous 8-element runs -> ds_load_b128)
    v16bf qf[3];
    const ushort_t* sQrow = &sQ[wave * 16 + r][0];
#pragma unroll
    for (int c = 0; c < 3; ++c) {
#pragma unroll
        for (int j = 0; j < 16; ++j) {
            int kl = half * 16 + j;
            qf[c][j] = asbf(sQrow[c * 32 + kl]);
        }
    }

    float m_r[8], l_r[8];
#pragma unroll
    for (int v = 0; v < 8; ++v) { m_r[v] = -1e30f; l_r[v] = 0.0f; }
    v8f o_acc[5];
#pragma unroll
    for (int t = 0; t < 5; ++t) o_acc[t] = zero8();

    for (int kt = 0; kt < SEQ / 32; ++kt) {
        const int key0 = kt * 32;
        // stage K tile via async: 32 rows x 9 chunks
        for (int idx = tid; idx < 32 * 9; idx += 128) {
            int rr = idx / 9, cc = (idx % 9) * 8;
            async_copy_b128(&sK[rr][cc], Kg + (size_t)(key0 + rr) * HDIM + cc);
        }
        // stage V tile transposed: sVt[d][key]
#pragma unroll
        for (int it = 0; it < 9; ++it) {
            int idx = tid + it * 128;              // 32*36 dwords
            int rr = idx / 36, c = idx % 36;
            uint_t val = Vg[(size_t)(key0 + rr) * 36 + c];
            sVt[2 * c][rr]     = (ushort_t)(val & 0xFFFFu);
            sVt[2 * c + 1][rr] = (ushort_t)(val >> 16);
        }
        async_wait0();
        __syncthreads();

        // S = Q K^T (two 16-key sub-tiles, 3 K-chunks each)
        v8f s0 = zero8(), s1 = zero8();
#pragma unroll
        for (int c = 0; c < 3; ++c) {
            v16bf kf0, kf1;
            const ushort_t* k0p = &sK[r][c * 32 + half * 16];
            const ushort_t* k1p = &sK[16 + r][c * 32 + half * 16];
#pragma unroll
            for (int j = 0; j < 16; ++j) { kf0[j] = asbf(k0p[j]); kf1[j] = asbf(k1p[j]); }
            s0 = __builtin_amdgcn_wmma_f32_16x16x32_bf16(false, qf[c], false, kf0, (short)0, s0, false, false);
            s1 = __builtin_amdgcn_wmma_f32_16x16x32_bf16(false, qf[c], false, kf1, (short)0, s1, false, false);
        }

        // online softmax (per-row stats live in each 16-lane half)
        float alpha[8];
#pragma unroll
        for (int v = 0; v < 8; ++v) {
            float a0 = s0[v] * scale, a1 = s1[v] * scale;
            float mx = dpp_max16(fmaxf(a0, a1));
            float mnew = fmaxf(m_r[v], mx);
            alpha[v] = __expf(m_r[v] - mnew);
            m_r[v] = mnew;
            float p0 = __expf(a0 - mnew), p1 = __expf(a1 - mnew);
            float rs = dpp_sum16(p0 + p1);
            l_r[v] = l_r[v] * alpha[v] + rs;
            sP[wave][v + 8 * half][r]      = f2bf(p0);
            sP[wave][v + 8 * half][16 + r] = f2bf(p1);
        }
        __syncthreads();

        // P fragment + rescale O + P @ V
        v16bf pf;
        const ushort_t* sPr = &sP[wave][r][0];
#pragma unroll
        for (int j = 0; j < 16; ++j) {
            int kl = half * 16 + j;
            pf[j] = asbf(sPr[kl]);
        }
#pragma unroll
        for (int t = 0; t < 5; ++t) {
#pragma unroll
            for (int v = 0; v < 8; ++v) o_acc[t][v] *= alpha[v];
        }
#pragma unroll
        for (int t = 0; t < 5; ++t) {
            v16bf vf;
            const ushort_t* vp = &sVt[t * 16 + r][half * 16];
#pragma unroll
            for (int j = 0; j < 16; ++j) vf[j] = asbf(vp[j]);
            o_acc[t] = __builtin_amdgcn_wmma_f32_16x16x32_bf16(
                false, pf, false, vf, (short)0, o_acc[t], false, false);
        }
        __syncthreads();
    }

    // normalize + write (B, N, C) bf16, col = h*72 + d
    const int bidx = bh >> 4, hidx = bh & 15;
#pragma unroll
    for (int t = 0; t < 5; ++t) {
#pragma unroll
        for (int v = 0; v < 8; ++v) {
            int d = t * 16 + r;
            if (d < HDIM) {
                int row = q0 + wave * 16 + v + 8 * half;
                float val = o_acc[t][v] / l_r[v];
                O[((size_t)bidx * SEQ + row) * C_DIM + hidx * HDIM + d] = f2bf(val);
            }
        }
    }
}

// ---------------------------------------------------------------------------
extern "C" void kernel_launch(void* const* d_in, const int* in_sizes, int n_in,
                              void* d_out, int out_size, void* d_ws, size_t ws_size,
                              hipStream_t stream) {
    const float* x      = (const float*)d_in[0];
    const float* ln1_g  = (const float*)d_in[1];
    const float* ln1_b  = (const float*)d_in[2];
    const float* qkv_w  = (const float*)d_in[3];
    const float* qkv_b  = (const float*)d_in[4];
    const float* proj_w = (const float*)d_in[5];
    const float* proj_b = (const float*)d_in[6];
    const float* ln2_g  = (const float*)d_in[7];
    const float* ln2_b  = (const float*)d_in[8];
    const float* fc1_w  = (const float*)d_in[9];
    const float* fc1_b  = (const float*)d_in[10];
    const float* fc2_w  = (const float*)d_in[11];
    const float* fc2_b  = (const float*)d_in[12];
    float* out = (float*)d_out;

    char* ws = (char*)d_ws;
    ushort_t* h_bf = (ushort_t*)ws; ws += (size_t)NTOK * C_DIM * 2;
    ushort_t* wbuf = (ushort_t*)ws; ws += (size_t)C_DIM * INNER_DIM * 2;
    ushort_t* qbuf = (ushort_t*)ws; ws += (size_t)NTOK * C_DIM * 2;
    ushort_t* kbuf = (ushort_t*)ws; ws += (size_t)NTOK * C_DIM * 2;
    ushort_t* vbuf = (ushort_t*)ws; ws += (size_t)NTOK * C_DIM * 2;
    ushort_t* attn = (ushort_t*)ws; ws += (size_t)NTOK * C_DIM * 2;
    float*    x2   = (float*)ws;    ws += (size_t)NTOK * C_DIM * 4;
    ushort_t* g1   = (ushort_t*)ws; ws += (size_t)NTOK * INNER_DIM * 2;

    // 1. LN1 -> bf16
    ln_kernel<<<NTOK, 256, 0, stream>>>(x, ln1_g, ln1_b, h_bf);

    // 2. QKV = h @ qkv_w + b, scattered into Q/K/V (B,H,N,D) bf16
    {
        int n4 = C_DIM * N3C / 4;
        cvt_kernel<<<(n4 + 255) / 256, 256, 0, stream>>>(qkv_w, wbuf, n4);
        dim3 grid(NTOK / 64, N3C / 64);
        gemm_wmma<2><<<grid, 128, 0, stream>>>(h_bf, wbuf, qkv_b, nullptr,
                                               nullptr, nullptr, qbuf, kbuf, vbuf,
                                               C_DIM, N3C);
    }

    // 3. flash attention -> attn (B,N,C) bf16
    {
        dim3 grid(SEQ / 64, 256);
        attn_kernel<<<grid, 128, 0, stream>>>(qbuf, kbuf, vbuf, attn);
    }

    // 4. x2 = x + attn @ proj_w + proj_b
    {
        int n4 = C_DIM * C_DIM / 4;
        cvt_kernel<<<(n4 + 255) / 256, 256, 0, stream>>>(proj_w, wbuf, n4);
        dim3 grid(NTOK / 64, C_DIM / 64);
        gemm_wmma<0><<<grid, 128, 0, stream>>>(attn, wbuf, proj_b, x,
                                               x2, nullptr, nullptr, nullptr, nullptr,
                                               C_DIM, C_DIM);
    }

    // 5. LN2 -> bf16 (reuse h_bf)
    ln_kernel<<<NTOK, 256, 0, stream>>>(x2, ln2_g, ln2_b, h_bf);

    // 6. g1 = gelu(h2 @ fc1_w + fc1_b) bf16
    {
        int n4 = C_DIM * INNER_DIM / 4;
        cvt_kernel<<<(n4 + 255) / 256, 256, 0, stream>>>(fc1_w, wbuf, n4);
        dim3 grid(NTOK / 64, INNER_DIM / 64);
        gemm_wmma<1><<<grid, 128, 0, stream>>>(h_bf, wbuf, fc1_b, nullptr,
                                               nullptr, g1, nullptr, nullptr, nullptr,
                                               C_DIM, INNER_DIM);
    }

    // 7. out = x2 + g1 @ fc2_w + fc2_b
    {
        int n4 = INNER_DIM * C_DIM / 4;
        cvt_kernel<<<(n4 + 255) / 256, 256, 0, stream>>>(fc2_w, wbuf, n4);
        dim3 grid(NTOK / 64, C_DIM / 64);
        gemm_wmma<0><<<grid, 128, 0, stream>>>(g1, wbuf, fc2_b, x2,
                                               out, nullptr, nullptr, nullptr, nullptr,
                                               INNER_DIM, C_DIM);
    }
}
